// DeformableTransformer_24773371363681
// MI455X (gfx1250) — compile-verified, matrix operand order checked by reading, not run on hardware
//
#include <hip/hip_runtime.h>
#include <hip/hip_bf16.h>

// ---------------- problem constants (compile-time, from reference) ----------
#define D      256
#define HEADS  8
#define LVLS   4
#define PTS    4
#define NLAYERS 6
#define DFF    1024
#define BATCH  4
#define NTOK   5440                 // 64*64 + 32*32 + 16*16 + 8*8
#define MROWS  (BATCH * NTOK)       // 21760 rows for all GEMMs (divisible by 128)

typedef __attribute__((ext_vector_type(16))) __bf16 v16bf;
typedef __attribute__((ext_vector_type(2)))  __bf16 bf16x2;
typedef __attribute__((ext_vector_type(2)))  float  f32x2;
typedef __attribute__((ext_vector_type(8)))  float  v8f;

// f32 -> bf16 via hardware convert (RNE); packed pair form lets the backend
// select cvt_pk when available instead of a long bit-twiddling sequence.
__device__ __forceinline__ unsigned short f2bf(float f) {
  union { __bf16 h; unsigned short u; } c;
  c.h = (__bf16)f;
  return c.u;
}
__device__ __forceinline__ unsigned packbf2(float lo, float hi) {
  f32x2 v; v.x = lo; v.y = hi;
  union { bf16x2 b; unsigned u; } c;
  c.b = __builtin_convertvector(v, bf16x2);
  return c.u;
}

// ---------------------------------------------------------------------------
// GEMM: C[M x Kout] = (A (+A2)) [M x K] * W [K x Kout] + bias, optional ReLU.
// Workgroup = 256 threads = 8 waves -> 128x64 C tile; wave = 32x32 (2x2 WMMA
// tiles of 16x16). K staged in LDS as bf16 in 32-wide steps, double-buffered
// and software-pipelined (float4 global prefetch into regs overlapping WMMA).
// LDS row stride 42 ushorts (21 dwords, coprime with 64 banks) ->
// conflict-free 32-bit fragment reads and A-stage stores.
// ---------------------------------------------------------------------------
#define LSTR 42

template<bool HASA2>
__global__ __launch_bounds__(256)
void gemm_bf16_wmma(const float* __restrict__ A, const float* __restrict__ A2,
                    const float* __restrict__ W, const float* __restrict__ bias,
                    float* __restrict__ C, int K, int Kout, int relu)
{
  __shared__ __align__(16) unsigned short sA[2][128 * LSTR];
  __shared__ __align__(16) unsigned short sB[2][64 * LSTR];

  const int t     = threadIdx.x;
  const int lane  = t & 31;
  const int wave  = t >> 5;
  const int waveM = wave >> 1;          // 0..3  (32-row strip)
  const int waveN = wave & 1;           // 0..1  (32-col strip)
  const int rowBase = blockIdx.x * 128;
  const int colBase = blockIdx.y * 64;
  const int l16  = lane & 15;
  const int half = lane >> 4;

  // staging coordinates (vectorized float4)
  const int ka  = (t & 7) * 4;          // A: k in {0,4,...,28}
  const int ma  = t >> 3;               // A: row 0..31 (+32 per it)
  const int nb  = (t & 15) * 4;         // B: col in {0,4,...,60}
  const int kb0 = t >> 4;               // B: k 0..15 (+16 per it)

  v8f acc[2][2];
#pragma unroll
  for (int i = 0; i < 2; ++i)
#pragma unroll
    for (int j = 0; j < 2; ++j)
#pragma unroll
      for (int r = 0; r < 8; ++r) acc[i][j][r] = 0.0f;

  float4 aR[4], a2R[4], bR[2];

  auto loadTile = [&](int ks) {
#pragma unroll
    for (int it = 0; it < 4; ++it) {
      const int m = ma + it * 32;
      aR[it] = *(const float4*)(A + (rowBase + m) * K + ks + ka);
      if (HASA2) a2R[it] = *(const float4*)(A2 + (rowBase + m) * K + ks + ka);
    }
#pragma unroll
    for (int it = 0; it < 2; ++it) {
      const int k = kb0 + it * 16;
      bR[it] = *(const float4*)(W + (ks + k) * Kout + colBase + nb);
    }
  };

  auto storeTile = [&](int buf) {
#pragma unroll
    for (int it = 0; it < 4; ++it) {
      const int m = ma + it * 32;
      float x0 = aR[it].x, x1 = aR[it].y, x2 = aR[it].z, x3 = aR[it].w;
      if (HASA2) { x0 += a2R[it].x; x1 += a2R[it].y; x2 += a2R[it].z; x3 += a2R[it].w; }
      unsigned* p = (unsigned*)&sA[buf][m * LSTR + ka];   // dword-aligned (ka mult of 4)
      p[0] = packbf2(x0, x1);
      p[1] = packbf2(x2, x3);
    }
#pragma unroll
    for (int it = 0; it < 2; ++it) {
      const int k = kb0 + it * 16;
      sB[buf][(nb + 0) * LSTR + k] = f2bf(bR[it].x);
      sB[buf][(nb + 1) * LSTR + k] = f2bf(bR[it].y);
      sB[buf][(nb + 2) * LSTR + k] = f2bf(bR[it].z);
      sB[buf][(nb + 3) * LSTR + k] = f2bf(bR[it].w);
    }
  };

  const int nsteps = K >> 5;

  loadTile(0);
  storeTile(0);
  __syncthreads();

  union Frag { v16bf v; unsigned u[8]; };

  for (int s = 0; s < nsteps; ++s) {
    const int cur = s & 1;
    const bool more = (s + 1 < nsteps);
    if (more) loadTile((s + 1) << 5);   // global prefetch overlaps WMMA below

    // ---- build fragments per ISA 16-bit 16x32 layout ----------------------
    Frag fa[2], fb[2];
#pragma unroll
    for (int ss = 0; ss < 2; ++ss) {
      const unsigned short* rowA = &sA[cur][(waveM * 32 + ss * 16 + l16) * LSTR];
      const unsigned short* rowB = &sB[cur][(waveN * 32 + ss * 16 + l16) * LSTR];
#pragma unroll
      for (int j = 0; j < 8; ++j) {
        const int kb = ((j < 4) ? (2 * j) : (16 + 2 * (j - 4))) + half * 8;
        fa[ss].u[j] = *(const unsigned*)(rowA + kb);
        fb[ss].u[j] = *(const unsigned*)(rowB + kb);
      }
    }

#pragma unroll
    for (int i = 0; i < 2; ++i)
#pragma unroll
      for (int j = 0; j < 2; ++j)
        acc[i][j] = __builtin_amdgcn_wmma_f32_16x16x32_bf16(
            false, fa[i].v, false, fb[j].v, (short)0, acc[i][j], false, false);

    if (more) {
      storeTile(1 - cur);               // writes buffer fenced by previous barrier
      __syncthreads();
    }
  }

  // ---- epilogue: bias (+ReLU), f32 store ----------------------------------
#pragma unroll
  for (int i = 0; i < 2; ++i) {
#pragma unroll
    for (int j = 0; j < 2; ++j) {
      const int col = colBase + waveN * 32 + j * 16 + l16;
      const float bv = bias[col];
      const int m0 = rowBase + waveM * 32 + i * 16 + half * 8;
#pragma unroll
      for (int r = 0; r < 8; ++r) {
        float v = acc[i][j][r] + bv;
        if (relu) v = fmaxf(v, 0.0f);
        C[(m0 + r) * Kout + col] = v;
      }
    }
  }
}

// ---------------------------------------------------------------------------
// Deformable sampling: one wave32 per (b, q, h); lane = channel within head.
// value is (B, N, D) f32 -> 128B coalesced per corner read; everything is
// L2-resident (whole working set < 192MB).
// ---------------------------------------------------------------------------
__global__ __launch_bounds__(256)
void msda_sample(const float* __restrict__ value, const float* __restrict__ off,
                 const float* __restrict__ attnlog,
                 const float* __restrict__ valid_ratios,
                 float* __restrict__ outbuf)
{
  const int lane = threadIdx.x & 31;
  const int wid  = blockIdx.x * 8 + (threadIdx.x >> 5);   // (b*N + q)*H + h
  const int h  = wid % HEADS;
  const int bq = wid / HEADS;
  const int q  = bq % NTOK;
  const int b  = bq / NTOK;

  const int lvlH[4]   = {64, 32, 16, 8};
  const int lvlW[4]   = {64, 32, 16, 8};
  const int lstart[4] = {0, 4096, 5120, 5376};

  // query's own level -> normalized reference point
  const int lq = (q >= 5376) ? 3 : (q >= 5120) ? 2 : (q >= 4096) ? 1 : 0;
  const int local = q - lstart[lq];
  const int Wq = lvlW[lq], Hq = lvlH[lq];
  const int yq = local / Wq, xq = local % Wq;
  const float rbx = (xq + 0.5f) / (valid_ratios[(b * LVLS + lq) * 2 + 0] * (float)Wq);
  const float rby = (yq + 0.5f) / (valid_ratios[(b * LVLS + lq) * 2 + 1] * (float)Hq);

  // softmax over LVLS*PTS = 16 logits (uniform across the wave)
  const float* al = attnlog + (b * NTOK + q) * (HEADS * LVLS * PTS) + h * 16;
  float lg[16], mx = -1e30f;
#pragma unroll
  for (int p = 0; p < 16; ++p) { lg[p] = al[p]; mx = fmaxf(mx, lg[p]); }
  float ssum = 0.0f;
#pragma unroll
  for (int p = 0; p < 16; ++p) { lg[p] = __expf(lg[p] - mx); ssum += lg[p]; }
  const float inv = 1.0f / ssum;

  const float* ofp = off + (bq * (HEADS * LVLS * PTS * 2)) + h * 32; // l*8+p*2+c
  float acc = 0.0f;

#pragma unroll
  for (int l = 0; l < LVLS; ++l) {
    const int Wl = lvlW[l], Hl = lvlH[l];
    const float refx = rbx * valid_ratios[(b * LVLS + l) * 2 + 0];
    const float refy = rby * valid_ratios[(b * LVLS + l) * 2 + 1];
    const float* vbase = value + (b * NTOK + lstart[l]) * D + h * 32 + lane;
#pragma unroll
    for (int p = 0; p < PTS; ++p) {
      const float x = refx * (float)Wl + ofp[(l * 4 + p) * 2 + 0] - 0.5f;
      const float y = refy * (float)Hl + ofp[(l * 4 + p) * 2 + 1] - 0.5f;
      const float x0f = floorf(x), y0f = floorf(y);
      const float fx = x - x0f, fy = y - y0f;
      const int x0 = (int)x0f, y0 = (int)y0f;
      const float w = lg[l * 4 + p] * inv;
      const float w00 = w * (1.f - fx) * (1.f - fy);
      const float w10 = w * fx * (1.f - fy);
      const float w01 = w * (1.f - fx) * fy;
      const float w11 = w * fx * fy;
      if (x0 >= 0     && x0 < Wl     && y0 >= 0     && y0 < Hl)
        acc += w00 * vbase[(y0 * Wl + x0) * D];
      if (x0 + 1 >= 0 && x0 + 1 < Wl && y0 >= 0     && y0 < Hl)
        acc += w10 * vbase[(y0 * Wl + x0 + 1) * D];
      if (x0 >= 0     && x0 < Wl     && y0 + 1 >= 0 && y0 + 1 < Hl)
        acc += w01 * vbase[((y0 + 1) * Wl + x0) * D];
      if (x0 + 1 >= 0 && x0 + 1 < Wl && y0 + 1 >= 0 && y0 + 1 < Hl)
        acc += w11 * vbase[((y0 + 1) * Wl + x0 + 1) * D];
    }
  }
  outbuf[bq * D + h * 32 + lane] = acc;
}

// ---------------------------------------------------------------------------
// out = LayerNorm(out + x2) * g + be  (row = 256 elements, block per row)
// ---------------------------------------------------------------------------
__global__ __launch_bounds__(256)
void add_layernorm(const float* __restrict__ x2, const float* __restrict__ g,
                   const float* __restrict__ be, float* __restrict__ out)
{
  __shared__ float red[256];
  const int row = blockIdx.x;
  const int t   = threadIdx.x;
  const int idx = row * D + t;

  const float x = out[idx] + x2[idx];
  red[t] = x;
  __syncthreads();
  for (int s = 128; s > 0; s >>= 1) {
    if (t < s) red[t] += red[t + s];
    __syncthreads();
  }
  const float mean = red[0] * (1.0f / D);
  __syncthreads();

  const float d = x - mean;
  red[t] = d * d;
  __syncthreads();
  for (int s = 128; s > 0; s >>= 1) {
    if (t < s) red[t] += red[t + s];
    __syncthreads();
  }
  const float var = red[0] * (1.0f / D);
  const float rs  = rsqrtf(var + 1e-5f);
  out[idx] = d * rs * g[t] + be[t];
}

// ---------------------------------------------------------------------------
extern "C" void kernel_launch(void* const* d_in, const int* in_sizes, int n_in,
                              void* d_out, int out_size, void* d_ws, size_t ws_size,
                              hipStream_t stream)
{
  (void)in_sizes; (void)n_in; (void)out_size; (void)ws_size;
  const float* src   = (const float*)d_in[0];
  const float* pos   = (const float*)d_in[1];
  const float* valid = (const float*)d_in[2];
  // d_in[3] spatial_shapes, d_in[4] level_start_index: compile-time constants
  const float* W_off  = (const float*)d_in[5];
  const float* b_off  = (const float*)d_in[6];
  const float* W_attn = (const float*)d_in[7];
  const float* b_attn = (const float*)d_in[8];
  const float* W_val  = (const float*)d_in[9];
  const float* b_val  = (const float*)d_in[10];
  const float* W_out  = (const float*)d_in[11];
  const float* b_out  = (const float*)d_in[12];
  const float* g1     = (const float*)d_in[13];
  const float* be1    = (const float*)d_in[14];
  const float* W1     = (const float*)d_in[15];
  const float* bl1    = (const float*)d_in[16];
  const float* W2     = (const float*)d_in[17];
  const float* bl2    = (const float*)d_in[18];
  const float* g2     = (const float*)d_in[19];
  const float* be2    = (const float*)d_in[20];

  float* out = (float*)d_out;
  float* ws  = (float*)d_ws;

  // workspace partition (floats)
  const size_t nBND = (size_t)MROWS * D;      // 5,570,560
  float* wsValue = ws;
  float* wsOff   = wsValue + nBND;
  float* wsAttn  = wsOff   + nBND;
  float* wsMsda  = wsAttn  + (size_t)MROWS * 128;
  float* wsTmp   = wsMsda  + nBND;
  float* wsFfn   = wsTmp   + nBND;            // MROWS * DFF

  hipMemcpyAsync(out, src, nBND * sizeof(float), hipMemcpyDeviceToDevice, stream);

  const dim3 blk(256);
  const int MB = MROWS / 128;                 // 170 row-blocks

  for (int i = 0; i < NLAYERS; ++i) {
    // value = out @ W_val + b_val
    gemm_bf16_wmma<false><<<dim3(MB, D / 64), blk, 0, stream>>>(
        out, nullptr, W_val + i * D * D, b_val + i * D, wsValue, D, D, 0);
    // off = (out + pos) @ W_off + b_off        (Kout = 256)
    gemm_bf16_wmma<true><<<dim3(MB, 256 / 64), blk, 0, stream>>>(
        out, pos, W_off + i * D * 256, b_off + i * 256, wsOff, D, 256, 0);
    // attn logits = (out + pos) @ W_attn + b_attn  (Kout = 128)
    gemm_bf16_wmma<true><<<dim3(MB, 128 / 64), blk, 0, stream>>>(
        out, pos, W_attn + i * D * 128, b_attn + i * 128, wsAttn, D, 128, 0);
    // deformable sampling -> msda (B,N,D)
    msda_sample<<<dim3(MROWS * HEADS / 8), blk, 0, stream>>>(
        wsValue, wsOff, wsAttn, valid, wsMsda);
    // x2 = msda @ W_out + b_out
    gemm_bf16_wmma<false><<<dim3(MB, D / 64), blk, 0, stream>>>(
        wsMsda, nullptr, W_out + i * D * D, b_out + i * D, wsTmp, D, D, 0);
    // out = LN(out + x2)
    add_layernorm<<<dim3(MROWS), blk, 0, stream>>>(wsTmp, g1 + i * D, be1 + i * D, out);
    // ffn_h = relu(out @ W1 + bl1)
    gemm_bf16_wmma<false><<<dim3(MB, DFF / 64), blk, 0, stream>>>(
        out, nullptr, W1 + (size_t)i * D * DFF, bl1 + i * DFF, wsFfn, D, DFF, 1);
    // x2 = ffn_h @ W2 + bl2
    gemm_bf16_wmma<false><<<dim3(MB, 4), blk, 0, stream>>>(
        wsFfn, nullptr, W2 + (size_t)i * DFF * D, bl2 + i * D, wsTmp, DFF, D, 0);
    // out = LN(out + ffn)
    add_layernorm<<<dim3(MROWS), blk, 0, stream>>>(wsTmp, g2 + i * D, be2 + i * D, out);
  }
}
